// Block_5214090297662
// MI455X (gfx1250) — matrix operand order repaired and checked by reading, not compiled, not run on hardware
//
#include <hip/hip_runtime.h>
#include <math.h>

// ---- problem constants ----
#define Bb 2
#define Ss 2048
#define Dd 1024
#define Hh 16
#define HD 64
#define EPSV 1e-5f

typedef __bf16 v16bf __attribute__((ext_vector_type(16)));
typedef __bf16 bf16x2 __attribute__((ext_vector_type(2)));
typedef float  v8f   __attribute__((ext_vector_type(8)));

__device__ __forceinline__ int lane_id() { return threadIdx.x & 31; }

// cross-lane xor shuffle via ds_bpermute (wave32-safe, no warpSize macros)
__device__ __forceinline__ float lane_xor_f(float v, int mask) {
  int i = __builtin_bit_cast(int, v);
  int r = __builtin_amdgcn_ds_bpermute((lane_id() ^ mask) << 2, i);
  return __builtin_bit_cast(float, r);
}

__device__ __forceinline__ float wave_reduce_add(float v) {
#pragma unroll
  for (int m = 1; m <= 16; m <<= 1) v += lane_xor_f(v, m);
  return v;
}

// async global->LDS copy, 16B per lane (gfx1250 GLOBAL_LOAD_ASYNC_TO_LDS_B128, ASYNCcnt)
__device__ __forceinline__ void async_copy_b128(uint32_t lds_off, const void* gaddr) {
  asm volatile("global_load_async_to_lds_b128 %0, %1, off"
               :
               : "v"(lds_off), "v"((uint64_t)(uintptr_t)gaddr)
               : "memory");
}
__device__ __forceinline__ void wait_async0() {
  asm volatile("s_wait_asynccnt 0x0" ::: "memory");
}

// ---- WMMA fragment loaders (CDNA5 ISA 7.12.2 layouts) ----
// A 16x32 bf16: lane L -> row m=L&15, half=L/16; VGPR v<4: K=2v+8*half, v>=4: K=16+2(v-4)+8*half
__device__ __forceinline__ v16bf load_frag_a(const __bf16* base, int ld) {
  const int lane = lane_id();
  const int m = lane & 15, h = lane >> 4;
  const __bf16* row = base + m * ld + h * 8;
  v16bf f;
#pragma unroll
  for (int v = 0; v < 4; ++v) {
    bf16x2 lo = *(const bf16x2*)(row + 2 * v);
    bf16x2 hi = *(const bf16x2*)(row + 16 + 2 * v);
    f[2 * v] = lo.x;     f[2 * v + 1] = lo.y;
    f[8 + 2 * v] = hi.x; f[8 + 2 * v + 1] = hi.y;
  }
  return f;
}

// B 32x16 bf16 stored k-contiguous per column: element (k,n) at base[n*ld + k];
// lane L -> col n=L&15, half=L/16; v<4: K=2v+8*half, v>=4: K=16+2(v-4)+8*half
__device__ __forceinline__ v16bf load_frag_b(const __bf16* base, int ld) {
  const int lane = lane_id();
  const int n = lane & 15, h = lane >> 4;
  const __bf16* row = base + n * ld + h * 8;
  v16bf f;
#pragma unroll
  for (int v = 0; v < 4; ++v) {
    bf16x2 lo = *(const bf16x2*)(row + 2 * v);
    bf16x2 hi = *(const bf16x2*)(row + 16 + 2 * v);
    f[2 * v] = lo.x;     f[2 * v + 1] = lo.y;
    f[8 + 2 * v] = hi.x; f[8 + 2 * v + 1] = hi.y;
  }
  return f;
}

__device__ __forceinline__ v8f wmma_bf16(v16bf a, v16bf b, v8f c) {
  return __builtin_amdgcn_wmma_f32_16x16x32_bf16(false, a, false, b, (short)0, c,
                                                 false, false);
}

__device__ __forceinline__ float gelu_f(float x) {
  const float k = 0.7978845608028654f;  // sqrt(2/pi)
  float t = tanhf(k * (x + 0.044715f * x * x * x));
  return 0.5f * x * (1.0f + t);
}

// ---- fp32 -> bf16 conversion ----
__global__ __launch_bounds__(256) void f32_to_bf16_kernel(const float* __restrict__ in,
                                                          __bf16* __restrict__ out, int n) {
  int i = blockIdx.x * 256 + threadIdx.x;
  if (i < n) out[i] = (__bf16)in[i];
}

// ---- LayerNorm (torch-style: Bessel var, eps added to std) fp32 -> bf16 ----
__global__ __launch_bounds__(256) void layernorm_bf16_kernel(const float* __restrict__ x,
                                                             const float* __restrict__ g,
                                                             const float* __restrict__ beta,
                                                             __bf16* __restrict__ out) {
  const int row = blockIdx.x;
  const float* xr = x + (size_t)row * Dd;
  const int tid = threadIdx.x;
  float vals[4];
  float lsum = 0.f;
#pragma unroll
  for (int i = 0; i < 4; ++i) { vals[i] = xr[tid + 256 * i]; lsum += vals[i]; }
  lsum = wave_reduce_add(lsum);
  __shared__ float red[8];
  __shared__ float stat[2];
  const int w = tid >> 5, lane = tid & 31;
  if (lane == 0) red[w] = lsum;
  __syncthreads();
  if (tid == 0) { float t = 0; for (int i = 0; i < 8; ++i) t += red[i]; stat[0] = t / Dd; }
  __syncthreads();
  const float u = stat[0];
  float ls2 = 0.f;
#pragma unroll
  for (int i = 0; i < 4; ++i) { float c = vals[i] - u; ls2 += c * c; }
  ls2 = wave_reduce_add(ls2);
  if (lane == 0) red[w] = ls2;
  __syncthreads();
  if (tid == 0) {
    float t = 0; for (int i = 0; i < 8; ++i) t += red[i];
    stat[1] = 1.0f / (sqrtf(t / (float)(Dd - 1)) + EPSV);
  }
  __syncthreads();
  const float inv = stat[1];
#pragma unroll
  for (int i = 0; i < 4; ++i) {
    int c = tid + 256 * i;
    out[(size_t)row * Dd + c] = (__bf16)(g[c] * ((vals[i] - u) * inv) + beta[c]);
  }
}

// ---- split qkv [M,3D] -> q,k [B,H,S,HD], vT [B,H,HD,S] ----
__global__ __launch_bounds__(256) void split_qkv_kernel(const __bf16* __restrict__ qkv,
                                                        __bf16* __restrict__ q,
                                                        __bf16* __restrict__ k,
                                                        __bf16* __restrict__ vt) {
  int idx = blockIdx.x * 256 + threadIdx.x;  // over B*S*D
  if (idx >= Bb * Ss * Dd) return;
  int d = idx & (HD - 1);
  int h = (idx >> 6) & (Hh - 1);
  int s = (idx >> 10) & (Ss - 1);
  int b = idx >> 21;
  size_t src = (size_t)(b * Ss + s) * (3 * Dd) + h * HD + d;
  size_t bh = (size_t)(b * Hh + h);
  q[(bh * Ss + s) * HD + d] = qkv[src];
  k[(bh * Ss + s) * HD + d] = qkv[src + Dd];
  vt[(bh * HD + d) * Ss + s] = qkv[src + 2 * Dd];
}

// ---- tiled bf16 WMMA GEMM: C[M,N] = A[M,K] @ B[K,N] + bias ----
// Macro tile 64x128, 8 waves, wave=32x32 (2x2 WMMA tiles) -> 4 wmma / wave / K-step.
// Double-buffered LDS; A tile staged via GLOBAL_LOAD_ASYNC_TO_LDS_B128 (ASYNCcnt),
// B tile staged via coalesced b128 loads + transposed LDS scatter, issued ahead of compute.
// mode 0: out_bf = acc+bias ; mode 1: out_bf = gelu(acc+bias) ; mode 2: out_f = acc+bias+resid
#define BM 64
#define BN 128
#define BK 32
#define LDA 40  // A LDS stride: 16B-aligned rows for async b128, conflict-free frag reads
#define LDB 34  // Bt LDS stride
__global__ __launch_bounds__(256) void gemm_bf16_kernel(
    const __bf16* __restrict__ A, const __bf16* __restrict__ B,
    const float* __restrict__ bias, const float* __restrict__ resid,
    __bf16* __restrict__ out_bf, float* __restrict__ out_f,
    int M, int N, int K, int mode) {
  __shared__ __bf16 As[2][BM][LDA];   // A tiles, row-major [m][k]
  __shared__ __bf16 Bts[2][BN][LDB];  // B tiles, transposed [n][k]

  const int tid = threadIdx.x;
  const int w = tid >> 5;
  const int m0 = blockIdx.y * BM;
  const int n0 = blockIdx.x * BN;
  const int wr = (w >> 2) * 32;  // wave row offset within tile (0 or 32)
  const int wc = (w & 3) * 32;   // wave col offset within tile (0..96)

  v8f acc00{}, acc01{}, acc10{}, acc11{};

  // staging index precompute
  const int arow = tid >> 2, acb = (tid & 3) * 8;   // A: 64 rows x 32 k, 16B/thread
  const int bkr = tid >> 3, bnb = (tid & 7) * 16;   // B: 32 k-rows x 128 n, 32B/thread

  const int nstep = K / BK;

  // --- prologue: stage step 0 into buffer 0 ---
  async_copy_b128((uint32_t)(uintptr_t)&As[0][arow][acb],
                  A + (size_t)(m0 + arow) * K + acb);
  {
    const __bf16* brow = B + (size_t)bkr * N + n0 + bnb;
    const uint4 r0 = *(const uint4*)(brow);
    const uint4 r1 = *(const uint4*)(brow + 8);
    const __bf16* e0 = (const __bf16*)&r0;
    const __bf16* e1 = (const __bf16*)&r1;
#pragma unroll
    for (int i = 0; i < 8; ++i) Bts[0][bnb + i][bkr] = e0[i];
#pragma unroll
    for (int i = 0; i < 8; ++i) Bts[0][bnb + 8 + i][bkr] = e1[i];
  }
  wait_async0();
  __syncthreads();

  for (int s = 0; s < nstep; ++s) {
    const int cur = s & 1, nxt = cur ^ 1;
    const int k0 = s * BK;
    const bool has_next = (s + 1) < nstep;

    uint4 nb0, nb1;
    if (has_next) {
      // issue next-step staging ahead of compute: async A -> LDS, B -> VGPRs
      async_copy_b128((uint32_t)(uintptr_t)&As[nxt][arow][acb],
                      A + (size_t)(m0 + arow) * K + k0 + BK + acb);
      const __bf16* brow = B + (size_t)(k0 + BK + bkr) * N + n0 + bnb;
      nb0 = *(const uint4*)(brow);
      nb1 = *(const uint4*)(brow + 8);
      if (s + 2 < nstep)
        __builtin_prefetch(B + (size_t)(k0 + 2 * BK + bkr) * N + n0 + bnb, 0, 1);
    }

    // compute current buffer
    const v16bf a0 = load_frag_a(&As[cur][wr][0], LDA);
    const v16bf a1 = load_frag_a(&As[cur][wr + 16][0], LDA);
    const v16bf b0 = load_frag_b(&Bts[cur][wc][0], LDB);
    const v16bf b1 = load_frag_b(&Bts[cur][wc + 16][0], LDB);
    acc00 = wmma_bf16(a0, b0, acc00);
    acc01 = wmma_bf16(a0, b1, acc01);
    acc10 = wmma_bf16(a1, b0, acc10);
    acc11 = wmma_bf16(a1, b1, acc11);

    if (has_next) {
      const __bf16* e0 = (const __bf16*)&nb0;
      const __bf16* e1 = (const __bf16*)&nb1;
#pragma unroll
      for (int i = 0; i < 8; ++i) Bts[nxt][bnb + i][bkr] = e0[i];
#pragma unroll
      for (int i = 0; i < 8; ++i) Bts[nxt][bnb + 8 + i][bkr] = e1[i];
    }
    wait_async0();
    __syncthreads();
  }

  const int lane = lane_id();
  const int nl = lane & 15, hf = lane >> 4;
  v8f accs[2][2] = {{acc00, acc01}, {acc10, acc11}};
#pragma unroll
  for (int i = 0; i < 2; ++i) {
#pragma unroll
    for (int j = 0; j < 2; ++j) {
      const int gn = n0 + wc + j * 16 + nl;
      const float bi = bias[gn];
#pragma unroll
      for (int r = 0; r < 8; ++r) {
        const int gm = m0 + wr + i * 16 + r + 8 * hf;
        float v = accs[i][j][r] + bi;
        size_t o = (size_t)gm * N + gn;
        if (mode == 2) {
          out_f[o] = v + resid[o];
        } else {
          if (mode == 1) v = gelu_f(v);
          out_bf[o] = (__bf16)v;
        }
      }
    }
  }
}

// ---- flash attention: 1 wave per 16-row Q tile, online softmax, 32-col K chunks ----
__global__ __launch_bounds__(128) void attn_kernel(const __bf16* __restrict__ Q,
                                                   const __bf16* __restrict__ Kt,
                                                   const __bf16* __restrict__ Vt,
                                                   __bf16* __restrict__ O) {
  __shared__ __bf16 Pb[4][16][32];  // per-wave P tile (bf16, A-layout friendly)
  const int w = threadIdx.x >> 5;
  const int lane = lane_id();
  const int tile = blockIdx.x * 4 + w;          // B*H*(S/16) = 4096 tiles
  const int qt = tile & (Ss / 16 - 1);          // 0..127
  const int h = (tile >> 7) & (Hh - 1);
  const int b = tile >> 11;
  const __bf16* Qh = Q + (size_t)(b * Hh + h) * Ss * HD;
  const __bf16* Kh = Kt + (size_t)(b * Hh + h) * Ss * HD;
  const __bf16* Vh = Vt + (size_t)(b * Hh + h) * HD * Ss;
  const int q0 = qt * 16;

  const v16bf qa0 = load_frag_a(Qh + (size_t)q0 * HD, HD);       // d 0..31
  const v16bf qa1 = load_frag_a(Qh + (size_t)q0 * HD + 32, HD);  // d 32..63

  v8f o0{}, o1{}, o2{}, o3{};
  float mrow[8], lrow[8];
#pragma unroll
  for (int r = 0; r < 8; ++r) { mrow[r] = -1e30f; lrow[r] = 0.f; }

  const int nl = lane & 15, hf = lane >> 4;
  const int kend = q0 + 16;  // causal: cols <= q0+15

  for (int kc = 0; kc < kend; kc += 32) {
    // scores: S = Q Kᵀ / 8 ; two 16x16 col tiles (n and n+16), HD=64 -> 2 k-steps each
    v8f s0{}, s1{};
    s0 = wmma_bf16(qa0, load_frag_b(Kh + (size_t)kc * HD, HD), s0);
    s0 = wmma_bf16(qa1, load_frag_b(Kh + (size_t)kc * HD + 32, HD), s0);
    s1 = wmma_bf16(qa0, load_frag_b(Kh + (size_t)(kc + 16) * HD, HD), s1);
    s1 = wmma_bf16(qa1, load_frag_b(Kh + (size_t)(kc + 16) * HD + 32, HD), s1);

#pragma unroll
    for (int r = 0; r < 8; ++r) {
      const int grow = q0 + r + 8 * hf;
      float v0 = s0[r] * 0.125f;
      float v1 = s1[r] * 0.125f;
      if (kc + nl > grow) v0 = -1e30f;
      if (kc + 16 + nl > grow) v1 = -1e30f;
      // chunk row-max over the 16-lane row group
      float cm = fmaxf(v0, v1);
      cm = fmaxf(cm, lane_xor_f(cm, 1));
      cm = fmaxf(cm, lane_xor_f(cm, 2));
      cm = fmaxf(cm, lane_xor_f(cm, 4));
      cm = fmaxf(cm, lane_xor_f(cm, 8));
      const float nm = fmaxf(mrow[r], cm);
      const float sc = __expf(mrow[r] - nm);
      const float p0 = __expf(v0 - nm);
      const float p1 = __expf(v1 - nm);
      float rs = p0 + p1;
      rs += lane_xor_f(rs, 1);
      rs += lane_xor_f(rs, 2);
      rs += lane_xor_f(rs, 4);
      rs += lane_xor_f(rs, 8);
      lrow[r] = lrow[r] * sc + rs;
      mrow[r] = nm;
      o0[r] *= sc; o1[r] *= sc; o2[r] *= sc; o3[r] *= sc;
      Pb[w][r + 8 * hf][nl] = (__bf16)p0;
      Pb[w][r + 8 * hf][nl + 16] = (__bf16)p1;
    }
    // wave-private LDS round trip: P (C-layout) -> A-fragment layout
    asm volatile("s_wait_dscnt 0" ::: "memory");
    const v16bf pa = load_frag_a(&Pb[w][0][0], 32);
    o0 = wmma_bf16(pa, load_frag_b(Vh + (size_t)0 * Ss + kc, Ss), o0);
    o1 = wmma_bf16(pa, load_frag_b(Vh + (size_t)16 * Ss + kc, Ss), o1);
    o2 = wmma_bf16(pa, load_frag_b(Vh + (size_t)32 * Ss + kc, Ss), o2);
    o3 = wmma_bf16(pa, load_frag_b(Vh + (size_t)48 * Ss + kc, Ss), o3);
  }

#pragma unroll
  for (int r = 0; r < 8; ++r) {
    const int m = r + 8 * hf;
    const float inv = 1.0f / lrow[r];
    size_t orow = (size_t)(b * Ss + q0 + m) * Dd + h * HD;
    O[orow + nl] = (__bf16)(o0[r] * inv);
    O[orow + 16 + nl] = (__bf16)(o1[r] * inv);
    O[orow + 32 + nl] = (__bf16)(o2[r] * inv);
    O[orow + 48 + nl] = (__bf16)(o3[r] * inv);
  }
}

// ---- host orchestration ----
extern "C" void kernel_launch(void* const* d_in, const int* in_sizes, int n_in,
                              void* d_out, int out_size, void* d_ws, size_t ws_size,
                              hipStream_t stream) {
  const float* x        = (const float*)d_in[0];
  const float* ln1_w    = (const float*)d_in[1];
  const float* ln1_b    = (const float*)d_in[2];
  const float* c_attn_w = (const float*)d_in[3];
  const float* c_attn_b = (const float*)d_in[4];
  const float* c_proj_w = (const float*)d_in[5];
  const float* c_proj_b = (const float*)d_in[6];
  const float* ln2_w    = (const float*)d_in[7];
  const float* ln2_b    = (const float*)d_in[8];
  const float* fc_w     = (const float*)d_in[9];
  const float* fc_b     = (const float*)d_in[10];
  const float* proj_w   = (const float*)d_in[11];
  const float* proj_b   = (const float*)d_in[12];
  float* out = (float*)d_out;

  const int M = Bb * Ss;  // 4096
  char* p = (char*)d_ws;
  auto alloc = [&](size_t bytes) {
    char* r = p;
    p += (bytes + 255) & ~(size_t)255;
    return r;
  };
  __bf16* wqkv = (__bf16*)alloc((size_t)Dd * 3 * Dd * 2);
  __bf16* wpa  = (__bf16*)alloc((size_t)Dd * Dd * 2);
  __bf16* wfc  = (__bf16*)alloc((size_t)Dd * 4 * Dd * 2);
  __bf16* wp2  = (__bf16*)alloc((size_t)4 * Dd * Dd * 2);
  __bf16* xln1 = (__bf16*)alloc((size_t)M * Dd * 2);
  __bf16* qkv  = (__bf16*)alloc((size_t)M * 3 * Dd * 2);
  __bf16* qh   = (__bf16*)alloc((size_t)M * Dd * 2);
  __bf16* kh   = (__bf16*)alloc((size_t)M * Dd * 2);
  __bf16* vT   = (__bf16*)alloc((size_t)M * Dd * 2);
  __bf16* attn = (__bf16*)alloc((size_t)M * Dd * 2);
  float*  x1   = (float*)alloc((size_t)M * Dd * 4);
  __bf16* xln2 = (__bf16*)alloc((size_t)M * Dd * 2);
  __bf16* hbuf = (__bf16*)alloc((size_t)M * 4 * Dd * 2);

  // 1) weights -> bf16
  {
    int n;
    n = Dd * 3 * Dd;
    f32_to_bf16_kernel<<<(n + 255) / 256, 256, 0, stream>>>(c_attn_w, wqkv, n);
    n = Dd * Dd;
    f32_to_bf16_kernel<<<(n + 255) / 256, 256, 0, stream>>>(c_proj_w, wpa, n);
    n = Dd * 4 * Dd;
    f32_to_bf16_kernel<<<(n + 255) / 256, 256, 0, stream>>>(fc_w, wfc, n);
    n = 4 * Dd * Dd;
    f32_to_bf16_kernel<<<(n + 255) / 256, 256, 0, stream>>>(proj_w, wp2, n);
  }
  // 2) ln1(x) -> bf16
  layernorm_bf16_kernel<<<M, 256, 0, stream>>>(x, ln1_w, ln1_b, xln1);
  // 3) qkv = xln1 @ Wqkv + b
  gemm_bf16_kernel<<<dim3(3 * Dd / BN, M / BM), 256, 0, stream>>>(
      xln1, wqkv, c_attn_b, nullptr, qkv, nullptr, M, 3 * Dd, Dd, 0);
  // 4) split heads (+ V transpose)
  split_qkv_kernel<<<(Bb * Ss * Dd) / 256, 256, 0, stream>>>(qkv, qh, kh, vT);
  // 5) attention (flash, causal)
  attn_kernel<<<(Bb * Hh * (Ss / 16)) / 4, 128, 0, stream>>>(qh, kh, vT, attn);
  // 6) x1 = x + attn @ Wproj + b   (fp32 residual stream)
  gemm_bf16_kernel<<<dim3(Dd / BN, M / BM), 256, 0, stream>>>(
      attn, wpa, c_proj_b, x, nullptr, x1, M, Dd, Dd, 2);
  // 7) ln2(x1) -> bf16
  layernorm_bf16_kernel<<<M, 256, 0, stream>>>(x1, ln2_w, ln2_b, xln2);
  // 8) h = gelu(xln2 @ Wfc + b)
  gemm_bf16_kernel<<<dim3(4 * Dd / BN, M / BM), 256, 0, stream>>>(
      xln2, wfc, fc_b, nullptr, hbuf, nullptr, M, 4 * Dd, Dd, 1);
  // 9) out = x1 + h @ Wp2 + b
  gemm_bf16_kernel<<<dim3(Dd / BN, M / BM), 256, 0, stream>>>(
      hbuf, wp2, proj_b, x1, nullptr, out, M, Dd, 4 * Dd, 2);
}
